// CompactPointMamba_19172734009552
// MI455X (gfx1250) — matrix operand order repaired and checked by reading, not compiled, not run on hardware
//
#include <hip/hip_runtime.h>
#include <hip/hip_bf16.h>
#include <math.h>

typedef _Float16 half_t;
typedef __attribute__((ext_vector_type(16))) _Float16 v16h;
typedef __attribute__((ext_vector_type(8)))  _Float16 v8h;
typedef __attribute__((ext_vector_type(8)))  float    v8f;

static constexpr int kB  = 4;      // batches
static constexpr int kN  = 2048;   // points per batch
static constexpr int kL  = 4096;   // tokens per batch (2 orderings)
static constexpr int kE  = 128;    // embed dim
static constexpr int kDI = 256;    // inner dim
static constexpr int kDC = 4;      // conv taps
static constexpr int kDR = 8;      // dt rank
static constexpr int kDS = 16;     // state dim
static constexpr int kNC = 40;     // classes
static constexpr int kM  = kB * kL; // 16384 tokens total

union V16u { v16h v; v8h h[2]; };

__device__ __forceinline__ float sigmoidf_(float x){ return 1.f/(1.f+__expf(-x)); }
__device__ __forceinline__ float siluf_(float x){ return x*sigmoidf_(x); }

// ---------------- f32 -> f16 convert ----------------
__global__ void k_cvt_f16(const float* __restrict__ s, half_t* __restrict__ d, int n){
  int i = blockIdx.x*blockDim.x + threadIdx.x;
  if (i < n) d[i] = (half_t)s[i];
}

// ---------------- point gather + embedding ----------------
__global__ void k_embed(const float* __restrict__ x, const int* __restrict__ oh,
                        const int* __restrict__ ot, const float* __restrict__ pe_w,
                        const float* __restrict__ pe_b, const float* __restrict__ gamma,
                        const float* __restrict__ beta, float* __restrict__ h){
  int idx = blockIdx.x*blockDim.x + threadIdx.x;
  if (idx >= kM*kE) return;
  int e   = idx & (kE-1);
  int tok = idx >> 7;
  int l   = tok & (kL-1);
  int b   = tok >> 12;
  int hs  = (l >= kN) ? 1 : 0;
  int li  = hs ? (l - kN) : l;
  int src = hs ? ot[b*kN + li] : oh[b*kN + li];
  const float* p = x + ((size_t)b*kN + src)*3;
  float v = p[0]*pe_w[e*3+0] + p[1]*pe_w[e*3+1] + p[2]*pe_w[e*3+2] + pe_b[e];
  v = v * gamma[hs*kE + e] + beta[hs*kE + e];
  h[idx] = v;
}

// ---------------- per-token LayerNorm -> f16 ----------------
__global__ void k_ln_f16(const float* __restrict__ h, const float* __restrict__ g,
                         const float* __restrict__ bb, half_t* __restrict__ out){
  __shared__ float red[kE];
  int tok = blockIdx.x, e = threadIdx.x;
  float v = h[(size_t)tok*kE + e];
  red[e] = v; __syncthreads();
  for (int s = kE/2; s > 0; s >>= 1){ if (e < s) red[e] += red[e+s]; __syncthreads(); }
  float m = red[0] * (1.f/kE); __syncthreads();
  float dv = v - m; red[e] = dv*dv; __syncthreads();
  for (int s = kE/2; s > 0; s >>= 1){ if (e < s) red[e] += red[e+s]; __syncthreads(); }
  float rs = rsqrtf(red[0]*(1.f/kE) + 1e-5f);
  out[(size_t)tok*kE + e] = (half_t)(dv*rs*g[e] + bb[e]);
}

// ---------------- WMMA fragment loaders ----------------
// A (MxK, 16-bit): lanes 0-15 row M=lane, K in {kb..kb+7, kb+16..kb+23}, kb=(lane>>4)*8
__device__ __forceinline__ v16h fragA(const half_t* __restrict__ row, int k0, int lane){
  int kb = (lane >> 4) * 8;
  V16u r;
  r.h[0] = *(const v8h*)(row + k0 + kb);
  r.h[1] = *(const v8h*)(row + k0 + kb + 16);
  return r.v;
}
// B (KxN, 16-bit): per lane col N=lane&15, K in {kb..kb+7, kb+16..kb+23}, kb=(lane>>4)*8.
// uz = x @ W.T with W row-major (N,K): B[k][n] = W[n][k] -> contiguous in W's row.
__device__ __forceinline__ v16h fragB(const half_t* __restrict__ row, int k0, int lane){
  int kb = (lane >> 4) * 8;
  V16u r;
  r.h[0] = *(const v8h*)(row + k0 + kb);
  r.h[1] = *(const v8h*)(row + k0 + kb + 16);
  return r.v;
}

// C[M,N] = A[M,K] @ W[N,K]^T (+ Cin). One wave computes a 16x(16*NT) strip:
// all KT A-fragments held in registers, swept against NT weight tiles.
// EDGE=true handles N not a multiple of 16 (zero B columns, guarded stores).
template<int KT, int NT, bool EDGE>
__global__ void k_gemm(const half_t* __restrict__ A, const half_t* __restrict__ W,
                       const float* __restrict__ Cin, float* __restrict__ Cout,
                       int ldC, int Nvalid){
  constexpr int K = KT*32;
  int lane = threadIdx.x & 31;
  int wave = threadIdx.x >> 5;
  int mt  = blockIdx.x*4 + wave;
  int nt0 = blockIdx.y*NT;
  int r   = lane & 15;
  int lh  = (lane >> 4) * 8;
  const half_t* Arow = A + (size_t)(mt*16 + r) * K;

  v16h afrag[KT];
  #pragma unroll
  for (int k = 0; k < KT; ++k) afrag[k] = fragA(Arow, k*32, lane);

  int cm = mt*16 + lh;   // C rows cm..cm+7 for this lane half

  #pragma unroll
  for (int j = 0; j < NT; ++j){
    int n = (nt0 + j)*16 + r;
    bool wok = !EDGE || (n < Nvalid);
    const half_t* Wrow = W + (size_t)(EDGE ? (wok ? n : 0) : n) * K;

    v8f acc;
    if (Cin){
      #pragma unroll
      for (int v = 0; v < 8; ++v) acc[v] = Cin[(size_t)(cm+v)*ldC + n];
    } else {
      #pragma unroll
      for (int v = 0; v < 8; ++v) acc[v] = 0.f;
    }
    #pragma unroll
    for (int k = 0; k < KT; ++k){
      v16h b = fragB(Wrow, k*32, lane);
      if (EDGE && !wok){
        #pragma unroll
        for (int q = 0; q < 16; ++q) b[q] = (_Float16)0.f;
      }
      acc = __builtin_amdgcn_wmma_f32_16x16x32_f16(false, afrag[k], false, b,
                                                   (short)0, acc, false, false);
    }
    if (!EDGE || n < Nvalid){
      #pragma unroll
      for (int v = 0; v < 8; ++v) Cout[(size_t)(cm+v)*ldC + n] = acc[v];
    }
  }
}

// ---------------- causal depthwise conv (DC=4) + SiLU ----------------
__global__ void k_conv(const float* __restrict__ uz, const float* __restrict__ cw,
                       const float* __restrict__ cb, float* __restrict__ uc,
                       half_t* __restrict__ uc16){
  int idx = blockIdx.x*blockDim.x + threadIdx.x;
  if (idx >= kM*kDI) return;
  int d   = idx & (kDI-1);
  int tok = idx >> 8;
  int l   = tok & (kL-1);
  int b   = tok >> 12;
  float s = cb[d];
  #pragma unroll
  for (int k = 0; k < kDC; ++k){
    int lk = l + k - (kDC-1);
    if (lk >= 0)
      s += uz[((size_t)((b<<12)+lk))*(2*kDI) + d] * cw[d*kDC + k];
  }
  s = siluf_(s);
  uc[idx]   = s;
  uc16[idx] = (half_t)s;
}

// ---------------- delta = softplus(dt @ dtw^T + dtb) ----------------
__global__ void k_dt(const float* __restrict__ xdbl, const float* __restrict__ dtw,
                     const float* __restrict__ dtb, float* __restrict__ delta){
  int idx = blockIdx.x*blockDim.x + threadIdx.x;
  if (idx >= kM*kDI) return;
  int d = idx & (kDI-1); int tok = idx >> 8;
  const float* xr = xdbl + (size_t)tok*(kDR + 2*kDS);
  float s = dtb[d];
  #pragma unroll
  for (int r = 0; r < kDR; ++r) s += xr[r] * dtw[d*kDR + r];
  delta[idx] = (s > 20.f) ? s : log1pf(__expf(s));
}

// ---------------- selective-state scan: thread per (b,d), 16 states in regs --------
__global__ void k_scan(const float* __restrict__ delta, const float* __restrict__ uc,
                       const float* __restrict__ xdbl, const float* __restrict__ Alog,
                       const float* __restrict__ Dp, float* __restrict__ y){
  int idx = blockIdx.x*blockDim.x + threadIdx.x;
  if (idx >= kB*kDI) return;
  int d = idx & (kDI-1);
  int b = idx >> 8;
  float A[kDS], st[kDS];
  #pragma unroll
  for (int s = 0; s < kDS; ++s){ A[s] = -__expf(Alog[d*kDS + s]); st[s] = 0.f; }
  float Dv = Dp[d];
  for (int t = 0; t < kL; ++t){
    size_t tok = (size_t)b*kL + t;
    float dl = delta[tok*kDI + d];
    float ut = uc[tok*kDI + d];
    float du = dl * ut;
    const float* Bv = xdbl + tok*(kDR + 2*kDS) + kDR;
    const float* Cv = Bv + kDS;
    float yv = Dv * ut;
    #pragma unroll
    for (int s = 0; s < kDS; ++s){
      st[s] = __expf(dl*A[s])*st[s] + du*Bv[s];
      yv += st[s]*Cv[s];
    }
    y[tok*kDI + d] = yv;
  }
}

// ---------------- gating: y16 = y * silu(z) ----------------
__global__ void k_gate(const float* __restrict__ y, const float* __restrict__ uz,
                       half_t* __restrict__ y16){
  int idx = blockIdx.x*blockDim.x + threadIdx.x;
  if (idx >= kM*kDI) return;
  int d = idx & (kDI-1); int tok = idx >> 8;
  float z = uz[(size_t)tok*(2*kDI) + kDI + d];
  y16[idx] = (half_t)(y[idx] * siluf_(z));
}

// ---------------- final LN stats per token ----------------
__global__ void k_stats(const float* __restrict__ h, float* __restrict__ mo,
                        float* __restrict__ ro){
  __shared__ float red[kE];
  int tok = blockIdx.x, e = threadIdx.x;
  float v = h[(size_t)tok*kE + e];
  red[e] = v; __syncthreads();
  for (int s = kE/2; s > 0; s >>= 1){ if (e < s) red[e] += red[e+s]; __syncthreads(); }
  float m = red[0]*(1.f/kE); __syncthreads();
  float dv = v - m; red[e] = dv*dv; __syncthreads();
  for (int s = kE/2; s > 0; s >>= 1){ if (e < s) red[e] += red[e+s]; __syncthreads(); }
  if (e == 0){ mo[tok] = m; ro[tok] = rsqrtf(red[0]*(1.f/kE) + 1e-5f); }
}

// ---------------- mean over tokens of LN'ed h ----------------
__global__ void k_acc(const float* __restrict__ h, const float* __restrict__ m,
                      const float* __restrict__ r, const float* __restrict__ g,
                      const float* __restrict__ bb, float* __restrict__ acc){
  int idx = blockIdx.x*blockDim.x + threadIdx.x;
  if (idx >= kB*kE) return;
  int e = idx & (kE-1); int b = idx >> 7;
  float s = 0.f;
  for (int t = 0; t < kL; ++t){
    size_t tok = (size_t)b*kL + t;
    s += (h[tok*kE + e] - m[tok]) * r[tok];
  }
  acc[idx] = (s*(1.f/kL))*g[e] + bb[e];
}

// ---------------- classifier head ----------------
__global__ void k_fc(const float* __restrict__ acc, const float* __restrict__ fw,
                     const float* __restrict__ fb, float* __restrict__ out){
  int idx = blockIdx.x*blockDim.x + threadIdx.x;
  if (idx >= kB*kNC) return;
  int c = idx % kNC; int b = idx / kNC;
  float s = fb[c];
  for (int e = 0; e < kE; ++e) s += acc[b*kE + e] * fw[c*kE + e];
  out[idx] = s;
}

extern "C" void kernel_launch(void* const* d_in, const int* in_sizes, int n_in,
                              void* d_out, int out_size, void* d_ws, size_t ws_size,
                              hipStream_t stream) {
  (void)in_sizes; (void)n_in; (void)out_size; (void)ws_size;
  const float* x        = (const float*)d_in[0];
  const int*   oh       = (const int*)  d_in[1];
  const int*   ot       = (const int*)  d_in[2];
  const float* pe_w     = (const float*)d_in[3];
  const float* pe_b     = (const float*)d_in[4];
  const float* gamma    = (const float*)d_in[5];
  const float* beta     = (const float*)d_in[6];
  const float* ln_g     = (const float*)d_in[7];
  const float* ln_b     = (const float*)d_in[8];
  const float* inproj_w = (const float*)d_in[9];
  const float* conv_w   = (const float*)d_in[10];
  const float* conv_b   = (const float*)d_in[11];
  const float* xproj_w  = (const float*)d_in[12];
  const float* dtproj_w = (const float*)d_in[13];
  const float* dtproj_b = (const float*)d_in[14];
  const float* A_log    = (const float*)d_in[15];
  const float* Dp       = (const float*)d_in[16];
  const float* outproj_w= (const float*)d_in[17];
  const float* hn_g     = (const float*)d_in[18];
  const float* hn_b     = (const float*)d_in[19];
  const float* fc_w     = (const float*)d_in[20];
  const float* fc_b     = (const float*)d_in[21];
  float* out = (float*)d_out;

  // workspace carve-up
  char* ws = (char*)d_ws;
  size_t off = 0;
  auto carve = [&](size_t bytes)->char*{
    char* p = ws + off; off += (bytes + 255) & ~(size_t)255; return p;
  };
  float*  h     = (float*) carve((size_t)kM*kE*4);
  half_t* xn16  = (half_t*)carve((size_t)kM*kE*2);
  float*  uz    = (float*) carve((size_t)kM*2*kDI*4);
  float*  uc    = (float*) carve((size_t)kM*kDI*4);
  half_t* uc16  = (half_t*)carve((size_t)kM*kDI*2);
  float*  xdbl  = (float*) carve((size_t)kM*(kDR+2*kDS)*4);
  float*  delta = (float*) carve((size_t)kM*kDI*4);
  float*  yb    = (float*) carve((size_t)kM*kDI*4);
  half_t* y16   = (half_t*)carve((size_t)kM*kDI*2);
  half_t* w16in = (half_t*)carve((size_t)2*2*kDI*kE*2);        // (2, 512, 128)
  half_t* w16x  = (half_t*)carve((size_t)2*(kDR+2*kDS)*kDI*2); // (2, 40, 256)
  half_t* w16o  = (half_t*)carve((size_t)2*kE*kDI*2);          // (2, 128, 256)
  float*  statm = (float*) carve((size_t)kM*4);
  float*  statr = (float*) carve((size_t)kM*4);
  float*  accb  = (float*) carve((size_t)kB*kE*4);

  const int nIn = 2*2*kDI*kE;             // 131072
  const int nX  = 2*(kDR+2*kDS)*kDI;      // 20480
  const int nO  = 2*kE*kDI;               // 65536
  k_cvt_f16<<<(nIn+255)/256, 256, 0, stream>>>(inproj_w, w16in, nIn);
  k_cvt_f16<<<(nX +255)/256, 256, 0, stream>>>(xproj_w,  w16x,  nX);
  k_cvt_f16<<<(nO +255)/256, 256, 0, stream>>>(outproj_w, w16o, nO);

  k_embed<<<(kM*kE+255)/256, 256, 0, stream>>>(x, oh, ot, pe_w, pe_b, gamma, beta, h);

  const int mblk = kM/16/4;  // 256 blocks of 4 waves -> 1024 M-tiles
  for (int i = 0; i < 2; ++i){
    k_ln_f16<<<kM, kE, 0, stream>>>(h, ln_g + i*kE, ln_b + i*kE, xn16);
    // uz = xn @ inproj^T : (16384 x 512), K=128 (KT=4), 32 N-tiles, NT=4 -> grid.y=8
    k_gemm<4,4,false><<<dim3(mblk, 8), 128, 0, stream>>>(
        xn16, w16in + (size_t)i*2*kDI*kE, nullptr, uz, 2*kDI, 2*kDI);
    k_conv<<<(kM*kDI+255)/256, 256, 0, stream>>>(
        uz, conv_w + i*kDI*kDC, conv_b + i*kDI, uc, uc16);
    // xdbl = uc @ xproj^T : (16384 x 40), K=256 (KT=8), 3 N-tiles (edge-guarded)
    k_gemm<8,3,true><<<dim3(mblk, 1), 128, 0, stream>>>(
        uc16, w16x + (size_t)i*(kDR+2*kDS)*kDI, nullptr, xdbl,
        kDR+2*kDS, kDR+2*kDS);
    k_dt<<<(kM*kDI+255)/256, 256, 0, stream>>>(
        xdbl, dtproj_w + i*kDI*kDR, dtproj_b + i*kDI, delta);
    k_scan<<<(kB*kDI+255)/256, 256, 0, stream>>>(
        delta, uc, xdbl, A_log + i*kDI*kDS, Dp + i*kDI, yb);
    k_gate<<<(kM*kDI+255)/256, 256, 0, stream>>>(yb, uz, y16);
    // h += (y*silu(z)) @ outproj^T : (16384 x 128), K=256 (KT=8), 8 N-tiles, NT=4
    k_gemm<8,4,false><<<dim3(mblk, 2), 128, 0, stream>>>(
        y16, w16o + (size_t)i*kE*kDI, h, h, kE, kE);
  }

  k_stats<<<kM, kE, 0, stream>>>(h, statm, statr);
  k_acc<<<(kB*kE+255)/256, 256, 0, stream>>>(h, statm, statr, hn_g, hn_b, accb);
  k_fc<<<(kB*kNC+255)/256, 256, 0, stream>>>(accb, fc_w, fc_b, out);
}
